// GCN_78795470012585
// MI455X (gfx1250) — matrix operand order repaired and checked by reading, not compiled, not run on hardware
//
#include <hip/hip_runtime.h>
#include <hip/hip_bf16.h>
#include <math.h>

// GCN inference for MI455X (gfx1250, wave32).
// GEMMs: bf16 WMMA (v_wmma_f32_16x16x32_bf16), LDS-staged.
// Edge aggregation: wave-per-edge float4 gather + native fp32 global atomics.

typedef __attribute__((ext_vector_type(16))) __bf16 v16bf;
typedef __attribute__((ext_vector_type(8)))  float v8f;

__device__ __forceinline__ __bf16 to_bf16(float f) { return (__bf16)f; }

// ---------------------------------------------------------------------------
// Tiled GEMM: Out[nrows x COLS] = A[nrows x 128] * W[128 x COLS] (+ bias)
// One block = 256 threads = 8 waves; block covers 128 rows.
// COLS and HAS_BIAS are compile-time so interior blocks store with zero
// branching; the single ragged tail block (N % 128 != 0) takes a guarded path
// behind one block-uniform s_cbranch.
// LDS: W transposed (sW[n*128+k]) and 128 A rows, K contiguous per 32-K chunk,
// so each lane's 16-element WMMA fragment is a single contiguous 32B load.
// A-fragment lane layout (16-bit A 16x32, ISA 7.12.2):
//   lane half h = lane>>4; element j<8 -> K=8h+j ; j>=8 -> K=16+8h+(j-8)
// This code instead uses K=16h+j for both A and B: storage per 32-K chunk is [0..31].
// ---------------------------------------------------------------------------
template <int COLS, bool HAS_BIAS>
__global__ __launch_bounds__(256)
void GCN_gemm_bf16_wmma(const float* __restrict__ A, const float* __restrict__ W,
                        const float* __restrict__ bias, float* __restrict__ Out,
                        int nrows) {
  constexpr int NCT = (COLS + 15) / 16;    // col tiles
  constexpr int CP  = NCT * 16;            // padded cols staged in LDS
  __shared__ __attribute__((aligned(32))) __bf16 sA[128 * 128];
  __shared__ __attribute__((aligned(32))) __bf16 sW[CP * 128];
  const int t  = threadIdx.x;
  const int r0 = blockIdx.x * 128;

  // Stage W transposed: sW[n*128 + k] = W[k][n] (zero-pad n >= COLS).
  for (int idx = t; idx < CP * 128; idx += 256) {
    int n = idx >> 7, k = idx & 127;
    float v = (n < COLS) ? W[k * COLS + n] : 0.0f;
    sW[idx] = to_bf16(v);
  }
  // Stage 128 rows of A, bf16, K contiguous within each 32-chunk.
  for (int idx = t; idx < 128 * 128; idx += 256) {
    int r = idx >> 7, k = idx & 127;
    int row = r0 + r;
    float v = (row < nrows) ? A[(size_t)row * 128 + k] : 0.0f;
    int c = k & 31;
    int pos = c;
    sA[r * 128 + (k & ~31) + pos] = to_bf16(v);
  }
  __syncthreads();

  const int wid  = t >> 5;      // wave id -> 16-row strip
  const int lane = t & 31;
  const int half = lane >> 4;
  const int l16  = lane & 15;
  const bool fullRows = (r0 + 128 <= nrows);   // block-uniform

  // A fragments for this wave's rows: 4 K-chunks, held in VGPRs.
  v16bf afrag[4];
  const __bf16* arow = &sA[(wid * 16 + l16) * 128];
#pragma unroll
  for (int kc = 0; kc < 4; ++kc)
    afrag[kc] = *(const v16bf*)(arow + kc * 32 + half * 16);

  for (int ct = 0; ct < NCT; ++ct) {
    v8f acc = {};
#pragma unroll
    for (int kc = 0; kc < 4; ++kc) {
      // B fragment (32x16 bf16): element j -> K = kc*32 + 16*half + j, col = l16.
      v16bf bfrag = *(const v16bf*)(&sW[(ct * 16 + l16) * 128 + kc * 32 + half * 16]);
      acc = __builtin_amdgcn_wmma_f32_16x16x32_bf16(
          false, afrag[kc], false, bfrag, (short)0, acc, false, false);
    }
    // C/D layout: VGPR r -> row M = r + 8*half; N = lane%16.
    const int col = ct * 16 + l16;
    const bool colOK = (COLS % 16 == 0) ? true : (col < COLS);
    float* op = Out + (size_t)(r0 + wid * 16 + 8 * half) * COLS + col;
    if (fullRows) {
      if (colOK) {
        if constexpr (HAS_BIAS) {
          const float badd = bias[col];
#pragma unroll
          for (int r = 0; r < 8; ++r) op[(size_t)r * COLS] = acc[r] + badd;
        } else {
#pragma unroll
          for (int r = 0; r < 8; ++r) op[(size_t)r * COLS] = acc[r];
        }
      }
    } else {
      float badd = 0.0f;
      if (HAS_BIAS && colOK) badd = bias[col];
#pragma unroll
      for (int r = 0; r < 8; ++r) {
        int row = r0 + wid * 16 + 8 * half + r;
        if (row < nrows && colOK) {
          float v = acc[r];
          if constexpr (HAS_BIAS) v += badd;
          op[(size_t)r * COLS] = v;
        }
      }
    }
  }
}

// ---------------------------------------------------------------------------
// Degree / normalization helpers.
// ---------------------------------------------------------------------------
__global__ void GCN_init_ones(float* __restrict__ p, int n) {
  int i = blockIdx.x * 256 + threadIdx.x;
  if (i < n) p[i] = 1.0f;  // self-loop contributes 1 to every degree
}

__global__ void GCN_deg_accum(const int* __restrict__ dst, float* __restrict__ deg, int e) {
  int i = blockIdx.x * 256 + threadIdx.x;
  if (i < e)
    __hip_atomic_fetch_add(&deg[dst[i]], 1.0f, __ATOMIC_RELAXED, __HIP_MEMORY_SCOPE_AGENT);
}

__global__ void GCN_rsqrt_inplace(float* __restrict__ p, int n) {
  int i = blockIdx.x * 256 + threadIdx.x;
  if (i < n) p[i] = rsqrtf(p[i]);  // deg >= 1 always (self-loops)
}

// agg[i] = dinv[node]^2 * h[i]  (self-loop term; also zero-initializes agg)
__global__ void GCN_selfloop_scale(const float* __restrict__ dinv, const float* __restrict__ h,
                                   float* __restrict__ agg, int n) {
  int i = blockIdx.x * 256 + threadIdx.x;
  if (i < n * 128) {
    float d = dinv[i >> 7];
    agg[i] = d * d * h[i];
  }
}

// ---------------------------------------------------------------------------
// Edge aggregation: one wave32 per edge; float4 gather of h[src], fp32
// hardware atomics into agg[dst]. Prefetch a future edge's source row.
// ---------------------------------------------------------------------------
__global__ __launch_bounds__(256)
void GCN_edge_agg(const int* __restrict__ src, const int* __restrict__ dst,
                  const float* __restrict__ dinv, const float* __restrict__ h,
                  float* __restrict__ agg, int e) {
  int we   = (int)((blockIdx.x * blockDim.x + threadIdx.x) >> 5);
  int lane = threadIdx.x & 31;
  if (we >= e) return;

  if (we + 64 < e)
    __builtin_prefetch(h + (size_t)src[we + 64] * 128 + lane * 4, 0, 3);

  int s = src[we], d = dst[we];
  float norm = dinv[s] * dinv[d];
  float4 v = ((const float4*)(h + (size_t)s * 128))[lane];
  float* o = agg + (size_t)d * 128 + lane * 4;
  __hip_atomic_fetch_add(o + 0, v.x * norm, __ATOMIC_RELAXED, __HIP_MEMORY_SCOPE_AGENT);
  __hip_atomic_fetch_add(o + 1, v.y * norm, __ATOMIC_RELAXED, __HIP_MEMORY_SCOPE_AGENT);
  __hip_atomic_fetch_add(o + 2, v.z * norm, __ATOMIC_RELAXED, __HIP_MEMORY_SCOPE_AGENT);
  __hip_atomic_fetch_add(o + 3, v.w * norm, __ATOMIC_RELAXED, __HIP_MEMORY_SCOPE_AGENT);
}

// out = relu(in + bias[f]) (+ res)   -- f = i % 128
__global__ void GCN_bias_act(const float* __restrict__ in, const float* __restrict__ bias,
                             const float* __restrict__ res, float* __restrict__ out, int n) {
  int i = blockIdx.x * 256 + threadIdx.x;
  if (i < n * 128) {
    float v = fmaxf(in[i] + bias[i & 127], 0.0f);
    if (res) v += res[i];
    out[i] = v;
  }
}

// ---------------------------------------------------------------------------
// In-place log-softmax over 40 logits, one wave32 per node.
// ---------------------------------------------------------------------------
__global__ __launch_bounds__(256)
void GCN_log_softmax40(float* __restrict__ z, int n) {
  int node = (int)((blockIdx.x * blockDim.x + threadIdx.x) >> 5);
  int lane = threadIdx.x & 31;
  if (node >= n) return;
  float* row = z + (size_t)node * 40;
  float z0 = row[lane];                                         // lane < 32 < 40
  float z1 = (lane < 8) ? row[lane + 32] : -__builtin_inff();
  float m = fmaxf(z0, z1);
  for (int o = 16; o; o >>= 1) m = fmaxf(m, __shfl_xor(m, o, 32));
  float s = expf(z0 - m) + expf(z1 - m);                        // exp(-inf)=0
  for (int o = 16; o; o >>= 1) s += __shfl_xor(s, o, 32);
  float ls = m + logf(s);
  row[lane] = z0 - ls;
  if (lane < 8) row[lane + 32] = z1 - ls;
}

// ---------------------------------------------------------------------------
extern "C" void kernel_launch(void* const* d_in, const int* in_sizes, int n_in,
                              void* d_out, int out_size, void* d_ws, size_t ws_size,
                              hipStream_t stream) {
  const float* x  = (const float*)d_in[0];
  const int*   ei = (const int*)d_in[1];
  const float* w0 = (const float*)d_in[2];
  const float* b0 = (const float*)d_in[3];
  const float* w1 = (const float*)d_in[4];
  const float* b1 = (const float*)d_in[5];
  const float* wl = (const float*)d_in[6];
  const float* bl = (const float*)d_in[7];
  float* out = (float*)d_out;

  const int N = in_sizes[0] / 128;
  const int E = in_sizes[1] / 2;

  const int* src = ei;
  const int* dst = ei + E;

  float* ws = (float*)d_ws;
  const size_t NF = (size_t)N * 128;
  float* dinv = ws;
  float* hA = ws + (((size_t)N + 255) & ~(size_t)255);
  float* hB = hA + NF;
  float* hC = hB + NF;

  dim3 blk(256);
  int gN  = (N + 255) / 256;
  int gE  = (E + 255) / 256;
  int gNF = (int)((NF + 255) / 256);
  int gEw = (E + 7) / 8;       // 8 waves (edges) per block
  int gG  = (N + 127) / 128;   // GEMM: 128 rows per block
  int gLS = (N + 7) / 8;       // log-softmax: 8 nodes per block

  // D^{-1/2}
  GCN_init_ones<<<gN, blk, 0, stream>>>(dinv, N);
  GCN_deg_accum<<<gE, blk, 0, stream>>>(dst, dinv, E);
  GCN_rsqrt_inplace<<<gN, blk, 0, stream>>>(dinv, N);

  // Layer 0: hC = relu(Ahat * (x @ w0) + b0)
  GCN_gemm_bf16_wmma<128, false><<<gG, blk, 0, stream>>>(x, w0, nullptr, hA, N);
  GCN_selfloop_scale<<<gNF, blk, 0, stream>>>(dinv, hA, hB, N);
  GCN_edge_agg<<<gEw, blk, 0, stream>>>(src, dst, dinv, hA, hB, E);
  GCN_bias_act<<<gNF, blk, 0, stream>>>(hB, b0, nullptr, hC, N);

  // Layer 1 + residual: hA = relu(Ahat * (hC @ w1) + b1) + hC
  GCN_gemm_bf16_wmma<128, false><<<gG, blk, 0, stream>>>(hC, w1, nullptr, hA, N);
  GCN_selfloop_scale<<<gNF, blk, 0, stream>>>(dinv, hA, hB, N);
  GCN_edge_agg<<<gEw, blk, 0, stream>>>(src, dst, dinv, hA, hB, E);
  GCN_bias_act<<<gNF, blk, 0, stream>>>(hB, b1, hC, hA, N);

  // Classifier + log-softmax, in place in d_out (stride 40).
  GCN_gemm_bf16_wmma<40, true><<<gG, blk, 0, stream>>>(hA, wl, bl, out, N);
  GCN_log_softmax40<<<gLS, blk, 0, stream>>>(out, N);
}
